// _NLM_Layer_65455301591092
// MI455X (gfx1250) — matrix operand order repaired and checked by reading, not compiled, not run on hardware
//
#include <hip/hip_runtime.h>
#include <hip/hip_bf16.h>
#include <math.h>

typedef __attribute__((ext_vector_type(16))) _Float16 v16h;
typedef __attribute__((ext_vector_type(8)))  float    v8f;

#define BDIM 2
#define NDIM 512
#define P0C 16
#define P1C 32
#define P2C 32
#define HC 128
#define IN0C 80
#define IN1C 112
#define IN2C 128

#define W1PAD 136   // 272 B row: 16B-aligned, 68-dword stride -> conflict-free
#define HPAD  40    // 80 B row: 16B-aligned

__device__ __forceinline__ float sigmoidf_(float x) {
  return 1.0f / (1.0f + __expf(-x));
}

// A-fragment (16x32 f16): lane (hs*16+ln) holds row ln, k = [8hs..8hs+8) and [16+8hs..+8)
__device__ __forceinline__ v16h load_a_f32(const float* __restrict__ p0,
                                           const float* __restrict__ p1) {
  v16h a;
#pragma unroll
  for (int e = 0; e < 8; ++e) { a[e] = (_Float16)p0[e]; a[8 + e] = (_Float16)p1[e]; }
  return a;
}
__device__ __forceinline__ v16h load_a_lds(const _Float16* p0, const _Float16* p1) {
  v16h a;
#pragma unroll
  for (int e = 0; e < 8; ++e) { a[e] = p0[e]; a[8 + e] = p1[e]; }
  return a;
}
// B-fragment (32x16 f16): lane (hs*16+ln) holds col ln, k = [8hs..8hs+8) and [16+8hs..+8) (from [n][k] LDS)
__device__ __forceinline__ v16h load_b_lds(const _Float16* p) {
  v16h r;
#pragma unroll
  for (int e = 0; e < 8; ++e) { r[e] = p[e]; r[8 + e] = p[16 + e]; }
  return r;
}

// ---------------- out2: the big per-pair MLP via WMMA ----------------
__global__ __launch_bounds__(256) void k_out2(
    const float* __restrict__ x1, const float* __restrict__ x2,
    const float* __restrict__ W1, const float* __restrict__ b1,
    const float* __restrict__ W2, const float* __restrict__ b2,
    float* __restrict__ out2)
{
  __shared__ __align__(16) _Float16 sW1t[HC][W1PAD];   // W1 transposed: [n][k]
  __shared__ __align__(16) _Float16 sW2t[P2C][W1PAD];  // W2 transposed: [p][k]
  __shared__ float sB1[HC];
  __shared__ float sB2[P2C];
  __shared__ __align__(16) _Float16 sHid[8][16][HPAD]; // per-wave 16x32 bounce

  const int t = threadIdx.x;
  for (int idx = t; idx < HC * IN2C; idx += 256) {
    int n = idx >> 7, k = idx & 127;
    sW1t[n][k] = (_Float16)W1[k * HC + n];
  }
  for (int idx = t; idx < HC * P2C; idx += 256) {
    int p = idx & 31, k = idx >> 5;
    sW2t[p][k] = (_Float16)W2[k * P2C + p];
  }
  if (t < HC)  sB1[t] = b1[t];
  if (t < P2C) sB2[t] = b2[t];
  __syncthreads();

  const int wave = t >> 5;
  const int lane = t & 31;
  const int hs = lane >> 4;   // lane-half select
  const int ln = lane & 15;   // row (A/C) or col (B) index

  const int nblk = NDIM / 128;                // 4 j-blocks per (b,i)
  const int tile = blockIdx.x;                // 0 .. B*N*nblk-1
  const int jb = tile % nblk;
  const int i  = (tile / nblk) % NDIM;
  const int b  = tile / (nblk * NDIM);
  const int j0 = jb * 128 + wave * 16;        // this wave's 16 rows
  const int jrow = j0 + ln;                   // this lane's row

  // in2 row = [ x2[b,i,j,:32] | x1[b,i,:32] | x2[b,j,i,:32] | x1[b,j,:32] ]
  const float* s0 = x2 + (((size_t)b * NDIM + i) * NDIM + jrow) * P2C;
  const float* s1 = x1 + ((size_t)b * NDIM + i) * P1C;
  const float* s2 = x2 + (((size_t)b * NDIM + jrow) * NDIM + i) * P2C;
  const float* s3 = x1 + ((size_t)b * NDIM + jrow) * P1C;

  v16h A[4];
  A[0] = load_a_f32(s0 + 8 * hs, s0 + 16 + 8 * hs);
  A[1] = load_a_f32(s1 + 8 * hs, s1 + 16 + 8 * hs);
  A[2] = load_a_f32(s2 + 8 * hs, s2 + 16 + 8 * hs);
  A[3] = load_a_f32(s3 + 8 * hs, s3 + 16 + 8 * hs);

  v8f zero = {};
  v8f acc2[2];
  acc2[0] = zero; acc2[1] = zero;

#pragma unroll
  for (int kk2 = 0; kk2 < 4; ++kk2) {          // GEMM2 k-step == pair of GEMM1 n-tiles
#pragma unroll
    for (int half = 0; half < 2; ++half) {
      const int nt = 2 * kk2 + half;           // hidden column tile (16 cols)
      v8f c = {};
#pragma unroll
      for (int kk = 0; kk < 4; ++kk) {
        v16h bf = load_b_lds(&sW1t[16 * nt + ln][32 * kk + 8 * hs]);
        c = __builtin_amdgcn_wmma_f32_16x16x32_f16(false, A[kk], false, bf,
                                                   (short)0, c, false, false);
      }
      // bias + sigmoid, C-layout (lane holds rows 8hs+r at col 16nt+ln) -> LDS f16
      const float bias = sB1[16 * nt + ln];
#pragma unroll
      for (int r = 0; r < 8; ++r) {
        float v = sigmoidf_(c[r] + bias);
        sHid[wave][8 * hs + r][16 * half + ln] = (_Float16)v;
      }
    }
    // reload hidden 16x32 block in A-layout
    v16h a2 = load_a_lds(&sHid[wave][ln][8 * hs], &sHid[wave][ln][16 + 8 * hs]);
#pragma unroll
    for (int pt = 0; pt < 2; ++pt) {
      v16h bf2 = load_b_lds(&sW2t[16 * pt + ln][32 * kk2 + 8 * hs]);
      acc2[pt] = __builtin_amdgcn_wmma_f32_16x16x32_f16(false, a2, false, bf2,
                                                        (short)0, acc2[pt], false, false);
    }
  }

  // Non-temporal stores: out2 is write-once streaming data (64 MB); keep it from
  // evicting the x2 working set (64 MB, read twice in this kernel) out of the 192 MB L2.
  float* orow = out2 + (((size_t)b * NDIM + i) * NDIM + j0) * P2C;
#pragma unroll
  for (int pt = 0; pt < 2; ++pt) {
    const int p = 16 * pt + ln;
    const float bias = sB2[p];
#pragma unroll
    for (int r = 0; r < 8; ++r) {
      const int m = 8 * hs + r;
      __builtin_nontemporal_store(sigmoidf_(acc2[pt][r] + bias),
                                  &orow[(size_t)m * P2C + p]);
    }
  }
}

// ---------------- masked max/min over j of x2 ----------------
__global__ __launch_bounds__(256) void k_reduce2(
    const float* __restrict__ x2, float* __restrict__ rmax2, float* __restrict__ rmin2)
{
  __shared__ float sMx[8][P2C];
  __shared__ float sMn[8][P2C];
  const int t = threadIdx.x;
  const int p = t & 31;
  const int g = t >> 5;                         // 8 j-groups
  const int i = blockIdx.x % NDIM;
  const int b = blockIdx.x / NDIM;
  const float* base = x2 + (((size_t)b * NDIM + i) * NDIM) * P2C + p;
  float mx = -1e30f, mn = 1e30f;
  for (int j = g; j < NDIM; j += 8) {
    float v = base[(size_t)j * P2C];
    float vmx = (j == i) ? 0.0f : v;            // x2*m           (diag -> 0)
    float vmn = (j == i) ? 1.0f : v;            // x2*m + (1-m)   (diag -> 1)
    mx = fmaxf(mx, vmx);
    mn = fminf(mn, vmn);
  }
  sMx[g][p] = mx; sMn[g][p] = mn;
  __syncthreads();
  if (t < P2C) {
    float M = sMx[0][t], m2 = sMn[0][t];
    for (int g2 = 1; g2 < 8; ++g2) { M = fmaxf(M, sMx[g2][t]); m2 = fminf(m2, sMn[g2][t]); }
    rmax2[((size_t)b * NDIM + i) * P2C + t] = M;
    rmin2[((size_t)b * NDIM + i) * P2C + t] = m2;
  }
}

// ---------------- max/min over N of x1 ----------------
__global__ __launch_bounds__(64) void k_reduce1(
    const float* __restrict__ x1, float* __restrict__ rmax1, float* __restrict__ rmin1)
{
  const int t = threadIdx.x;                    // t = b*32 + p
  if (t >= BDIM * P1C) return;
  const int b = t >> 5, p = t & 31;
  const float* base = x1 + ((size_t)b * NDIM) * P1C + p;
  float mx = -1e30f, mn = 1e30f;
  for (int j = 0; j < NDIM; ++j) {
    float v = base[(size_t)j * P1C];
    mx = fmaxf(mx, v); mn = fminf(mn, v);
  }
  rmax1[t] = mx; rmin1[t] = mn;
}

// ---------------- out0: tiny MLP (B=2 rows) ----------------
__global__ __launch_bounds__(256) void k_out0(
    const float* __restrict__ x0, const float* __restrict__ rmax1, const float* __restrict__ rmin1,
    const float* __restrict__ W1, const float* __restrict__ b1,
    const float* __restrict__ W2, const float* __restrict__ b2,
    float* __restrict__ out0)
{
  __shared__ float sIn[BDIM][IN0C];
  __shared__ float sH[BDIM][HC];
  const int t = threadIdx.x;
  if (t < BDIM * IN0C) {
    int b = t / IN0C, f = t % IN0C;
    float v;
    if (f < P0C)              v = x0[b * P0C + f];
    else if (f < P0C + P1C)   v = rmax1[b * P1C + (f - P0C)];
    else                      v = rmin1[b * P1C + (f - P0C - P1C)];
    sIn[b][f] = v;
  }
  __syncthreads();
  {
    int b = t / HC, h = t % HC;                 // 256 threads = 2*128 exactly
    float acc = b1[h];
    for (int f = 0; f < IN0C; ++f) acc += sIn[b][f] * W1[f * HC + h];
    sH[b][h] = sigmoidf_(acc);
  }
  __syncthreads();
  if (t < BDIM * P0C) {
    int b = t / P0C, p = t % P0C;
    float acc = b2[p];
    for (int h = 0; h < HC; ++h) acc += sH[b][h] * W2[h * P0C + p];
    __builtin_nontemporal_store(sigmoidf_(acc), &out0[b * P0C + p]);
  }
}

// ---------------- out1: per-object MLP (B*N rows) ----------------
__global__ __launch_bounds__(128) void k_out1(
    const float* __restrict__ x0, const float* __restrict__ x1,
    const float* __restrict__ rmax2, const float* __restrict__ rmin2,
    const float* __restrict__ W1, const float* __restrict__ b1,
    const float* __restrict__ W2, const float* __restrict__ b2,
    float* __restrict__ out1)
{
  __shared__ float sIn[IN1C];
  __shared__ float sH[HC];
  const int t = threadIdx.x;
  const int i = blockIdx.x % NDIM;
  const int b = blockIdx.x / NDIM;
  if (t < IN1C) {
    float v;
    if (t < P1C)                        v = x1[((size_t)b * NDIM + i) * P1C + t];
    else if (t < P1C + P0C)             v = x0[b * P0C + (t - P1C)];
    else if (t < P1C + P0C + P2C)       v = rmax2[((size_t)b * NDIM + i) * P2C + (t - P1C - P0C)];
    else                                v = rmin2[((size_t)b * NDIM + i) * P2C + (t - P1C - P0C - P2C)];
    sIn[t] = v;
  }
  __syncthreads();
  {
    float acc = b1[t];
    for (int f = 0; f < IN1C; ++f) acc += sIn[f] * W1[f * HC + t];
    sH[t] = sigmoidf_(acc);
  }
  __syncthreads();
  if (t < P1C) {
    float acc = b2[t];
    for (int h = 0; h < HC; ++h) acc += sH[h] * W2[h * P1C + t];
    __builtin_nontemporal_store(sigmoidf_(acc),
                                &out1[((size_t)b * NDIM + i) * P1C + t]);
  }
}

extern "C" void kernel_launch(void* const* d_in, const int* in_sizes, int n_in,
                              void* d_out, int out_size, void* d_ws, size_t ws_size,
                              hipStream_t stream)
{
  (void)in_sizes; (void)n_in; (void)out_size; (void)ws_size;
  const float* x0   = (const float*)d_in[0];
  const float* x1   = (const float*)d_in[1];
  const float* x2   = (const float*)d_in[2];
  const float* W1_0 = (const float*)d_in[3];
  const float* b1_0 = (const float*)d_in[4];
  const float* W2_0 = (const float*)d_in[5];
  const float* b2_0 = (const float*)d_in[6];
  const float* W1_1 = (const float*)d_in[7];
  const float* b1_1 = (const float*)d_in[8];
  const float* W2_1 = (const float*)d_in[9];
  const float* b2_1 = (const float*)d_in[10];
  const float* W1_2 = (const float*)d_in[11];
  const float* b1_2 = (const float*)d_in[12];
  const float* W2_2 = (const float*)d_in[13];
  const float* b2_2 = (const float*)d_in[14];

  float* out0 = (float*)d_out;
  float* out1 = out0 + (size_t)BDIM * P0C;
  float* out2 = out1 + (size_t)BDIM * NDIM * P1C;

  float* ws    = (float*)d_ws;
  float* rmax2 = ws;
  float* rmin2 = rmax2 + (size_t)BDIM * NDIM * P2C;
  float* rmax1 = rmin2 + (size_t)BDIM * NDIM * P2C;
  float* rmin1 = rmax1 + (size_t)BDIM * P1C;

  k_reduce2<<<BDIM * NDIM, 256, 0, stream>>>(x2, rmax2, rmin2);
  k_reduce1<<<1, 64, 0, stream>>>(x1, rmax1, rmin1);
  k_out0<<<1, 256, 0, stream>>>(x0, rmax1, rmin1, W1_0, b1_0, W2_0, b2_0, out0);
  k_out1<<<BDIM * NDIM, 128, 0, stream>>>(x0, x1, rmax2, rmin2, W1_1, b1_1, W2_1, b2_1, out1);
  k_out2<<<BDIM * NDIM * (NDIM / 128), 256, 0, stream>>>(x1, x2, W1_2, b1_2, W2_2, b2_2, out2);
}
